// LTLfNet_83691732730117
// MI455X (gfx1250) — compile-verified, matrix operand order checked by reading, not run on hardware
//
#include <hip/hip_runtime.h>
#include <hip/hip_bf16.h>
#include <math.h>

#define BATCH 1024
#define D_    1001
#define DP    1024

typedef __attribute__((ext_vector_type(16))) __bf16 v16bf;
typedef __attribute__((ext_vector_type(8)))  float  v8f;

union Frag {
    uint4    u4[2];
    unsigned u[8];
    v16bf    v;
};

// Load a 16-element bf16 fragment slice for WMMA A/B layout:
// lane<16 -> K {k0..k0+7, k0+16..k0+23}; lane>=16 -> +8 on both halves.
__device__ __forceinline__ void load_frag(const __bf16* __restrict__ rowp,
                                          int k0, int half8, Frag& f) {
    f.u4[0] = *(const uint4*)(rowp + k0 + half8);
    f.u4[1] = *(const uint4*)(rowp + k0 + 16 + half8);
}

// ---------------------------------------------------------------------------
// Weight prep: transpose + pad + convert to bf16, N-major (row = output col n,
// contiguous over k).  mat 0: W_inv[0], 1: W_inv[1], 2: W_bin[:D], 3: W_bin[D:]
// ---------------------------------------------------------------------------
__global__ void __launch_bounds__(256)
prep_w_kernel(const float* __restrict__ W_inv, const float* __restrict__ W_bin,
              __bf16* __restrict__ Wt) {
    const int n   = blockIdx.x;   // output column 0..1023
    const int mat = blockIdx.y;   // 0..3
    __bf16* dst = Wt + ((size_t)mat << 20) + (size_t)n * DP;
    for (int k = threadIdx.x; k < DP; k += 256) {
        float v = 0.f;
        if (n < D_ && k < D_) {
            if (mat == 0)      v = W_inv[(size_t)k * D_ + n];
            else if (mat == 1) v = W_inv[(size_t)D_ * D_ + (size_t)k * D_ + n];
            else if (mat == 2) v = W_bin[(size_t)k * D_ + n];
            else               v = W_bin[(size_t)(k + D_) * D_ + n];
        }
        dst[k] = (__bf16)v;
    }
}

// ---------------------------------------------------------------------------
// Level 0: leaves are one-hot, so x@W is a row gather. One block per node.
// ---------------------------------------------------------------------------
__global__ void __launch_bounds__(256)
level0_kernel(const int* __restrict__ var_idx, const int* __restrict__ op_id,
              const float* __restrict__ W_inv, const float* __restrict__ b_inv,
              const float* __restrict__ W_bin, const float* __restrict__ b_bin,
              __bf16* __restrict__ xout) {
    const int node = blockIdx.x;          // 0 .. BATCH*32-1  (== b*32 + j)
    const int b = node >> 5, j = node & 31;
    const int vi = var_idx[(b << 6) + (j << 1)];
    const int vj = var_idx[(b << 6) + (j << 1) + 1];
    const int op = op_id[node];

    const float *r0, *r1, *bias;
    if (op == 0)      { r0 = W_inv + (size_t)vi * D_;
                        r1 = W_inv + (size_t)vj * D_;               bias = b_inv; }
    else if (op == 1) { r0 = W_inv + (size_t)D_ * D_ + (size_t)vi * D_;
                        r1 = W_inv + (size_t)D_ * D_ + (size_t)vj * D_; bias = b_inv + D_; }
    else              { r0 = W_bin + (size_t)vi * D_;
                        r1 = W_bin + (size_t)(D_ + vj) * D_;        bias = b_bin; }

    __bf16* dst = xout + (size_t)node * DP;
    for (int d = threadIdx.x; d < DP; d += 256) {
        float v = 0.f;
        if (d < D_) v = fmaxf(r0[d] + r1[d] + bias[d], 0.f);
        dst[d] = (__bf16)v;
    }
}

// ---------------------------------------------------------------------------
// Levels 1..5: per wave, one 16(M)x64(N) output strip; 3 candidate accumulators
// (inv0, inv1, bn).  sm = left + right via v_pk_add_bf16. 4 WMMAs per k-step
// per 16x16 sub-tile.  Select per-row op + bias + relu + bf16 store in epilogue.
// ---------------------------------------------------------------------------
__global__ void __launch_bounds__(256)
gemm_level_kernel(const __bf16* __restrict__ xin, __bf16* __restrict__ xout,
                  const __bf16* __restrict__ Wt,
                  const float* __restrict__ b_inv, const float* __restrict__ b_bin,
                  const int* __restrict__ op_id, int op_off, int n_out, int log2n) {
    const int lane  = threadIdx.x & 31;
    const int wave  = threadIdx.x >> 5;
    const int m0    = blockIdx.x << 4;                 // 16-row tile base
    const int nbase = (blockIdx.y << 9) + (wave << 6); // 64-col strip base
    const int half8 = (lane >> 4) << 3;
    const int hi8   = (lane >> 4) << 3;

    // A-row pointers for this lane (A-matrix row = m0 + (lane&15))
    const int mrow  = m0 + (lane & 15);
    const int btree = mrow >> log2n;
    const int j     = mrow & (n_out - 1);
    const int lrow  = (btree << (log2n + 1)) + (j << 1);
    const __bf16* lp = xin + (size_t)lrow * DP;
    const __bf16* rp = lp + DP;

    v8f acc0[4], acc1[4], accb[4];
    const v8f z = {};
#pragma unroll
    for (int t = 0; t < 4; ++t) { acc0[t] = z; acc1[t] = z; accb[t] = z; }

    const int coln0 = nbase + (lane & 15);

#pragma unroll 1
    for (int k0 = 0; k0 < DP; k0 += 32) {
        Frag fl, fr, fs;
        load_frag(lp, k0, half8, fl);
        load_frag(rp, k0, half8, fr);
#pragma unroll
        for (int i = 0; i < 8; ++i)
            asm volatile("v_pk_add_bf16 %0, %1, %2"
                         : "=v"(fs.u[i]) : "v"(fl.u[i]), "v"(fr.u[i]));
#pragma unroll
        for (int t = 0; t < 4; ++t) {
            const size_t col = (size_t)(coln0 + (t << 4));
            Frag w;
            load_frag(Wt + col * DP, k0, half8, w);                       // W_inv0^T
            acc0[t] = __builtin_amdgcn_wmma_f32_16x16x32_bf16(
                          false, fs.v, false, w.v, (short)0, acc0[t], false, false);
            load_frag(Wt + ((size_t)1 << 20) + col * DP, k0, half8, w);   // W_inv1^T
            acc1[t] = __builtin_amdgcn_wmma_f32_16x16x32_bf16(
                          false, fs.v, false, w.v, (short)0, acc1[t], false, false);
            load_frag(Wt + ((size_t)2 << 20) + col * DP, k0, half8, w);   // W_bin[:D]^T
            accb[t] = __builtin_amdgcn_wmma_f32_16x16x32_bf16(
                          false, fl.v, false, w.v, (short)0, accb[t], false, false);
            load_frag(Wt + ((size_t)3 << 20) + col * DP, k0, half8, w);   // W_bin[D:]^T
            accb[t] = __builtin_amdgcn_wmma_f32_16x16x32_bf16(
                          false, fr.v, false, w.v, (short)0, accb[t], false, false);
        }
    }

    // per-row op codes (C layout: VGPR i -> row m0 + i + 8*(lane>=16))
    int opv[8];
#pragma unroll
    for (int i = 0; i < 8; ++i)
        opv[i] = op_id[op_off + m0 + i + hi8];

#pragma unroll
    for (int t = 0; t < 4; ++t) {
        const int col = coln0 + (t << 4);
        const float b0 = (col < D_) ? b_inv[col]       : 0.f;
        const float b1 = (col < D_) ? b_inv[D_ + col]  : 0.f;
        const float bb = (col < D_) ? b_bin[col]       : 0.f;
#pragma unroll
        for (int i = 0; i < 8; ++i) {
            const int row = m0 + i + hi8;
            float v;
            if (opv[i] == 0)      v = acc0[t][i] + b0;
            else if (opv[i] == 1) v = acc1[t][i] + b1;
            else                  v = accb[t][i] + bb;
            v = fmaxf(v, 0.f);
            xout[(size_t)row * DP + col] = (__bf16)v;
        }
    }
}

// ---------------------------------------------------------------------------
// Head: h = relu(root @ W_u[uop] + b_u[uop]); logits = h @ W_out + b_out;
// log_softmax over 2 classes.  One block per tree.
// ---------------------------------------------------------------------------
__global__ void __launch_bounds__(256)
head_kernel(const __bf16* __restrict__ xroot, const int* __restrict__ uop_id,
            const float* __restrict__ W_u, const float* __restrict__ b_u,
            const float* __restrict__ W_out, const float* __restrict__ b_out,
            float* __restrict__ out) {
    __shared__ float root_s[DP];
    __shared__ float red0[256], red1[256];
    const int b = blockIdx.x, t = threadIdx.x;

    for (int d = t; d < DP; d += 256)
        root_s[d] = (float)xroot[(size_t)b * DP + d];
    __syncthreads();

    const int k = uop_id[b];
    const float* Wu = W_u + (size_t)k * D_ * D_;
    float acc[4] = {0.f, 0.f, 0.f, 0.f};
    for (int d = 0; d < D_; ++d) {
        const float r = root_s[d];
        const float* wrow = Wu + (size_t)d * D_;
#pragma unroll
        for (int s = 0; s < 4; ++s) {
            const int e = t + s * 256;
            if (e < D_) acc[s] = fmaf(r, wrow[e], acc[s]);
        }
    }
    float p0 = 0.f, p1 = 0.f;
#pragma unroll
    for (int s = 0; s < 4; ++s) {
        const int e = t + s * 256;
        if (e < D_) {
            const float h = fmaxf(acc[s] + b_u[(size_t)k * D_ + e], 0.f);
            p0 = fmaf(h, W_out[(size_t)e * 2 + 0], p0);
            p1 = fmaf(h, W_out[(size_t)e * 2 + 1], p1);
        }
    }
    red0[t] = p0; red1[t] = p1;
    __syncthreads();
    for (int s = 128; s > 0; s >>= 1) {
        if (t < s) { red0[t] += red0[t + s]; red1[t] += red1[t + s]; }
        __syncthreads();
    }
    if (t == 0) {
        const float l0 = red0[0] + b_out[0];
        const float l1 = red1[0] + b_out[1];
        const float m  = fmaxf(l0, l1);
        const float lse = m + logf(expf(l0 - m) + expf(l1 - m));
        out[b * 2 + 0] = l0 - lse;
        out[b * 2 + 1] = l1 - lse;
    }
}

// ---------------------------------------------------------------------------
extern "C" void kernel_launch(void* const* d_in, const int* in_sizes, int n_in,
                              void* d_out, int out_size, void* d_ws, size_t ws_size,
                              hipStream_t stream) {
    const int*   var_idx = (const int*)d_in[0];
    const int*   op_id   = (const int*)d_in[1];
    const int*   uop_id  = (const int*)d_in[2];
    const float* W_inv   = (const float*)d_in[3];
    const float* b_inv   = (const float*)d_in[4];
    const float* W_bin   = (const float*)d_in[5];
    const float* b_bin   = (const float*)d_in[6];
    const float* W_u     = (const float*)d_in[7];
    const float* b_u     = (const float*)d_in[8];
    const float* W_out   = (const float*)d_in[9];
    const float* b_out   = (const float*)d_in[10];
    float* out = (float*)d_out;

    // Workspace layout (bf16): xbuf0 (32768 rows), xbuf1 (16384 rows), Wt (4 x 1024x1024)
    __bf16* xbuf0 = (__bf16*)d_ws;
    __bf16* xbuf1 = xbuf0 + (size_t)32768 * DP;
    __bf16* Wt    = xbuf1 + (size_t)16384 * DP;

    prep_w_kernel<<<dim3(1024, 4), 256, 0, stream>>>(W_inv, W_bin, Wt);
    level0_kernel<<<dim3(BATCH * 32), 256, 0, stream>>>(var_idx, op_id, W_inv, b_inv,
                                                        W_bin, b_bin, xbuf0);

    __bf16* cur = xbuf0;
    __bf16* nxt = xbuf1;
    int off = BATCH * 32;
    for (int l = 1; l < 6; ++l) {
        const int n     = 64 >> (l + 1);   // 16, 8, 4, 2, 1
        const int log2n = 5 - l;
        const int M     = BATCH * n;
        gemm_level_kernel<<<dim3(M / 16, 2), 256, 0, stream>>>(
            cur, nxt, Wt, b_inv, b_bin, op_id, off, n, log2n);
        off += BATCH * n;
        __bf16* tmp = cur; cur = nxt; nxt = tmp;
    }

    head_kernel<<<dim3(BATCH), 256, 0, stream>>>(cur, uop_id, W_u, b_u, W_out, b_out, out);
}
